// DetectNet_35991825940546
// MI455X (gfx1250) — matrix operand order repaired and checked by reading, not compiled, not run on hardware
//
#include <hip/hip_runtime.h>

#define HH 2048
#define WW 2048
#define BIGL (HH*WW)
#define AS 40            // sAct row stride in halves: 80B = 16B-aligned, 20-bank stride

typedef __attribute__((ext_vector_type(16))) _Float16 v16h;
typedef __attribute__((ext_vector_type(8)))  _Float16 v8h;
typedef __attribute__((ext_vector_type(8)))  float    v8f;

#if __has_builtin(__builtin_amdgcn_global_load_async_to_lds_b128) && \
    __has_builtin(__builtin_amdgcn_s_wait_asynccnt)
#define HAS_ASYNC_LDS 1
typedef int v4i_ __attribute__((vector_size(4 * sizeof(int))));
typedef __attribute__((address_space(1))) v4i_ g4_t;   // global (printed as __device__)
typedef __attribute__((address_space(3))) v4i_ l4_t;   // LDS
#else
#define HAS_ASYNC_LDS 0
#endif

__device__ __forceinline__ int iminl(int a, int b) { return a < b ? a : b; }
__device__ __forceinline__ int imaxl(int a, int b) { return a > b ? a : b; }

// ---------------------------------------------------------------------------
// Weight prep: w2 (O=32,I=32,3,3) f32 -> f16 B-matrix layout [tap][ntile][n][k]
// ---------------------------------------------------------------------------
__global__ void k_prep_w2(const float* __restrict__ w2, _Float16* __restrict__ wB) {
    int i = blockIdx.x * 256 + threadIdx.x;
    if (i >= 9 * 32 * 32) return;
    int k  = i & 31;          // input channel
    int n  = (i >> 5) & 15;
    int nt = (i >> 9) & 1;
    int t  = i >> 10;         // tap 0..8
    int oc = nt * 16 + n;
    wB[i] = (_Float16)w2[(oc * 32 + k) * 9 + t];
}

// ---------------------------------------------------------------------------
// Fused cascade: conv1(1->32,3x3,SAME)+relu  ->  conv2(32->32,3x3,SAME)+relu
// via WMMA f16 -> conv3(1x1)+ (sigmoid>0.5 == logit>0) -> label init
// One 16x16 output tile per block, 256 threads (8 waves).
// Weights arrive over the async global->LDS path (ASYNCcnt) when available.
// ---------------------------------------------------------------------------
__global__ __launch_bounds__(256)
void k_cascade(const float* __restrict__ x, const float* __restrict__ w1,
               const float* __restrict__ w3, const _Float16* __restrict__ wB,
               int* __restrict__ lab)
{
    __shared__ float    sIn[20 * 20];         // input tile, origin (-2,-2)
    __shared__ float    sW1[288];
    __shared__ float    sW3[32];
    __shared__ _Float16 sAct[18 * 18 * AS];   // conv1 out, [pos][ch], origin (-1,-1)
    __shared__ _Float16 sW[288 * 32];         // B fragments [tap][nt][n][k]
    __shared__ _Float16 sC2[256 * 32];        // conv2 out [pos][ch]

    const int tid = threadIdx.x;
    const int tx0 = blockIdx.x * 16;
    const int ty0 = blockIdx.y * 16;

#if HAS_ASYNC_LDS
    // 18432 B of B-fragments: 1152 async b128 lane-transfers, no VGPR round-trip.
    for (int i = tid; i < 1152; i += 256) {
        __builtin_amdgcn_global_load_async_to_lds_b128(
            (g4_t*)(const void*)(wB + (size_t)i * 8),
            (l4_t*)(void*)(sW + (size_t)i * 8),
            0, 0);
    }
#else
    for (int i = tid; i < 9216; i += 256) sW[i] = wB[i];
#endif

    for (int i = tid; i < 400; i += 256) {
        int iy = i / 20, ix = i - iy * 20;
        int gy = ty0 + iy - 2, gx = tx0 + ix - 2;
        sIn[i] = (gy >= 0 && gy < HH && gx >= 0 && gx < WW) ? x[gy * WW + gx] : 0.f;
    }
    for (int i = tid; i < 288; i += 256) sW1[i] = w1[i];
    if (tid < 32) sW3[tid] = w3[tid];

#if HAS_ASYNC_LDS
    __builtin_amdgcn_s_wait_asynccnt(0);   // wave's async transfers visible in LDS
#endif
    __syncthreads();

    // conv1 + relu -> f16 activations (zero outside image: independent SAME pads)
    for (int o = tid; o < 18 * 18 * 32; o += 256) {
        int c = o & 31, p = o >> 5;
        int ay = p / 18, ax = p - ay * 18;
        int gy = ty0 + ay - 1, gx = tx0 + ax - 1;
        float acc = 0.f;
        if (gy >= 0 && gy < HH && gx >= 0 && gx < WW) {
            const float* wp = &sW1[c * 9];
            #pragma unroll
            for (int ky = 0; ky < 3; ++ky)
                #pragma unroll
                for (int kx = 0; kx < 3; ++kx)
                    acc += sIn[(ay + ky) * 20 + ax + kx] * wp[ky * 3 + kx];
            acc = fmaxf(acc, 0.f);
        }
        sAct[p * AS + c] = (_Float16)acc;
    }
    __syncthreads();

    // conv2 via WMMA: M-tile = one output row (16 px), N in 2 tiles of 16 oc,
    // K = 32 in-ch per tap, 9 taps accumulated in f32.
    const int lane = tid & 31;
    const int wv   = tid >> 5;       // wave 0..7 -> output rows 2w, 2w+1
    const int lr   = lane & 15;
    const int hi   = lane >> 4;
    const int kb   = hi * 8;         // A-fragment K base: runs {0..7,16..23}/{8..15,24..31}

    v8f acc[2][2] = {};
    #pragma unroll
    for (int t = 0; t < 9; ++t) {
        const int dy = t / 3, dx = t - dy * 3;
        const v8h* bp0 = (const v8h*)(sW + ((t * 2 + 0) * 16 + lr) * 32 + hi * 8);
        const v8h* bp1 = (const v8h*)(sW + ((t * 2 + 1) * 16 + lr) * 32 + hi * 8);
        v16h b0 = __builtin_shufflevector(bp0[0], bp0[2], 0,1,2,3,4,5,6,7,8,9,10,11,12,13,14,15);
        v16h b1 = __builtin_shufflevector(bp1[0], bp1[2], 0,1,2,3,4,5,6,7,8,9,10,11,12,13,14,15);
        #pragma unroll
        for (int mi = 0; mi < 2; ++mi) {
            const int mt = wv * 2 + mi;
            const v8h* ap = (const v8h*)(sAct + ((mt + dy) * 18 + lr + dx) * AS + kb);
            v16h a = __builtin_shufflevector(ap[0], ap[2], 0,1,2,3,4,5,6,7,8,9,10,11,12,13,14,15);
            acc[mi][0] = __builtin_amdgcn_wmma_f32_16x16x32_f16(false, a, false, b0,
                                                                (short)0, acc[mi][0], false, false);
            acc[mi][1] = __builtin_amdgcn_wmma_f32_16x16x32_f16(false, a, false, b1,
                                                                (short)0, acc[mi][1], false, false);
        }
    }

    // D layout: VGPR r -> M = r + 8*hi, N = lane%16  ->  relu + f16 store [pos][ch]
    #pragma unroll
    for (int mi = 0; mi < 2; ++mi) {
        const int mt = wv * 2 + mi;
        #pragma unroll
        for (int r = 0; r < 8; ++r) {
            int p = mt * 16 + r + hi * 8;
            sC2[p * 32 + lr]      = (_Float16)fmaxf(acc[mi][0][r], 0.f);
            sC2[p * 32 + 16 + lr] = (_Float16)fmaxf(acc[mi][1][r], 0.f);
        }
    }
    __syncthreads();

    // conv3 (1x1) + sigmoid>0.5 <=> logit>0 ; fold mask into label init
    {
        float d = 0.f;
        #pragma unroll
        for (int c = 0; c < 32; ++c) d += (float)sC2[tid * 32 + c] * sW3[c];
        int gy = ty0 + (tid >> 4), gx = tx0 + (tid & 15);
        int idx = gy * WW + gx;
        lab[idx] = (d > 0.f) ? idx : BIGL;
    }
}

// ---------------------------------------------------------------------------
// One CC min-propagation sweep (mask invariant: lab != BIG)
// ---------------------------------------------------------------------------
__global__ __launch_bounds__(256)
void k_cc_step(const int* __restrict__ src, int* __restrict__ dst) {
    int p = blockIdx.x * 256 + threadIdx.x;
    __builtin_prefetch(src + p + 8 * WW, 0, 1);   // global_prefetch_b8
    int v = src[p];
    if (v == BIGL) { dst[p] = BIGL; return; }
    int y = p >> 11, xx = p & 2047;
    int up = (y > 0)      ? src[p - WW] : BIGL;
    int dn = (y < HH - 1) ? src[p + WW] : BIGL;
    int lf = (xx > 0)     ? src[p - 1]  : BIGL;
    int rt = (xx < WW - 1)? src[p + 1]  : BIGL;
    dst[p] = iminl(v, iminl(iminl(up, dn), iminl(lf, rt)));
}

__global__ void k_fill_i32(int* __restrict__ buf, int n, int val) {
    int i = blockIdx.x * 256 + threadIdx.x;
    if (i < n) buf[i] = val;
}

// histogram with per-thread run merging (labels are spatially coherent)
__global__ __launch_bounds__(256)
void k_histo(const int* __restrict__ lab, int* __restrict__ counts) {
    int base = (blockIdx.x * 256 + threadIdx.x) * 16;
    int cur = -1, cnt = 0;
    for (int j = 0; j < 16; ++j) {
        int v = lab[base + j];
        if (v == BIGL) continue;
        if (v == cur) { cnt++; }
        else { if (cnt) atomicAdd(&counts[cur], cnt); cur = v; cnt = 1; }
    }
    if (cnt) atomicAdd(&counts[cur], cnt);
}

// ---------------------------------------------------------------------------
// top-k via 16x (argmax + knock out).  key = (count+1)<<32 | (~idx) so ties
// pick the smallest index (matches jax top_k ordering).
// ---------------------------------------------------------------------------
__global__ __launch_bounds__(256)
void k_amax_a(const int* __restrict__ counts, unsigned long long* __restrict__ partials) {
    __shared__ unsigned long long red[256];
    int tid = threadIdx.x;
    int gid = blockIdx.x * 256 + tid;
    unsigned long long best = 0ull;
    for (int i = gid; i < BIGL + 1; i += 256 * 1024) {
        unsigned long long key =
            ((unsigned long long)(unsigned)(counts[i] + 1) << 32) | (unsigned)(0xFFFFFFFFu - i);
        best = key > best ? key : best;
    }
    red[tid] = best;
    __syncthreads();
    for (int s = 128; s > 0; s >>= 1) {
        if (tid < s) { unsigned long long o = red[tid + s]; if (o > red[tid]) red[tid] = o; }
        __syncthreads();
    }
    if (tid == 0) partials[blockIdx.x] = red[0];
}

__global__ __launch_bounds__(256)
void k_amax_b(const unsigned long long* __restrict__ partials, int* __restrict__ counts,
              int* __restrict__ lids, int* __restrict__ cnts, int k) {
    __shared__ unsigned long long red[256];
    int tid = threadIdx.x;
    unsigned long long best = 0ull;
    for (int i = tid; i < 1024; i += 256) {
        unsigned long long key = partials[i];
        best = key > best ? key : best;
    }
    red[tid] = best;
    __syncthreads();
    for (int s = 128; s > 0; s >>= 1) {
        if (tid < s) { unsigned long long o = red[tid + s]; if (o > red[tid]) red[tid] = o; }
        __syncthreads();
    }
    if (tid == 0) {
        unsigned long long b = red[0];
        int idx = (int)(0xFFFFFFFFu - (unsigned)(b & 0xFFFFFFFFull));
        int val = (int)((unsigned)(b >> 32)) - 1;
        lids[k] = idx;
        cnts[k] = val;
        counts[idx] = -1;   // knock out for next round
    }
}

// ---------------------------------------------------------------------------
// Bounding boxes (LDS pre-reduction, then 1 global atomic per block per field)
// ---------------------------------------------------------------------------
__global__ void k_bbox_init(int* __restrict__ bbox) {
    int i = threadIdx.x;
    if (i < 64) bbox[i] = (i & 1) ? -1 : 0x7FFFFFFF;
}

__global__ __launch_bounds__(256)
void k_bbox_scan(const int* __restrict__ lab, const int* __restrict__ lids, int* __restrict__ bbox) {
    __shared__ int slid[16];
    __shared__ int sb[64];
    int tid = threadIdx.x;
    if (tid < 16) slid[tid] = lids[tid];
    if (tid < 64) sb[tid] = (tid & 1) ? -1 : 0x7FFFFFFF;
    __syncthreads();
    int p = blockIdx.x * 256 + tid;
    int v = lab[p];
    if (v != BIGL) {
        int kk = -1;
        #pragma unroll
        for (int k = 0; k < 16; ++k) if (slid[k] == v) kk = k;
        if (kk >= 0) {
            int y = p >> 11, xx = p & 2047;
            atomicMin(&sb[kk * 4 + 0], y);
            atomicMax(&sb[kk * 4 + 1], y);
            atomicMin(&sb[kk * 4 + 2], xx);
            atomicMax(&sb[kk * 4 + 3], xx);
        }
    }
    __syncthreads();
    if (tid < 64) {
        int val = sb[tid];
        if (tid & 1) { if (val >= 0)          atomicMax(&bbox[tid], val); }
        else         { if (val != 0x7FFFFFFF) atomicMin(&bbox[tid], val); }
    }
}

// ---------------------------------------------------------------------------
// ROI bilinear resize to 227x227 (empty bbox -> full frame, like argmax(false)=0)
// ---------------------------------------------------------------------------
__global__ __launch_bounds__(256)
void k_roi(const float* __restrict__ img, const int* __restrict__ bbox, float* __restrict__ roi) {
    int k = blockIdx.x;
    int i = blockIdx.y * 256 + threadIdx.x;
    if (i >= 227 * 227) return;
    int y0 = bbox[k * 4 + 0], y1 = bbox[k * 4 + 1];
    int x0 = bbox[k * 4 + 2], x1 = bbox[k * 4 + 3];
    if (y1 < 0) { y0 = 0; y1 = HH - 1; }
    if (x1 < 0) { x0 = 0; x1 = WW - 1; }
    int oy = i / 227, ox = i - oy * 227;
    float gy = (float)y0 + ((float)oy / 226.f) * (float)(y1 - y0);
    float gx = (float)x0 + ((float)ox / 226.f) * (float)(x1 - x0);
    int yf = (int)floorf(gy); yf = iminl(imaxl(yf, 0), HH - 2);
    int xf = (int)floorf(gx); xf = iminl(imaxl(xf, 0), WW - 2);
    float fy = gy - (float)yf, fx = gx - (float)xf;
    const float* r0 = img + yf * WW + xf;
    float v00 = r0[0], v01 = r0[1], v10 = r0[WW], v11 = r0[WW + 1];
    roi[k * 227 * 227 + i] =
        (1.f - fy) * ((1.f - fx) * v00 + fx * v01) + fy * ((1.f - fx) * v10 + fx * v11);
}

// classifier conv1: 1->16ch, 7x7, stride4, VALID -> 56x56
__global__ __launch_bounds__(256)
void k_cls_conv1(const float* __restrict__ roi, const float* __restrict__ cw1, float* __restrict__ h1) {
    __shared__ float sw[16 * 49];
    for (int i = threadIdx.x; i < 784; i += 256) sw[i] = cw1[i];
    __syncthreads();
    int k = blockIdx.x;
    int o = blockIdx.y * 256 + threadIdx.x;   // oc*3136 + oy*56 + ox ; 50176 total
    int oc = o / 3136;
    int rem = o - oc * 3136;
    int oy = rem / 56, ox = rem - oy * 56;
    const float* rp = roi + k * 227 * 227 + (oy * 4) * 227 + ox * 4;
    float acc = 0.f;
    #pragma unroll
    for (int ky = 0; ky < 7; ++ky)
        #pragma unroll
        for (int kx = 0; kx < 7; ++kx)
            acc += rp[ky * 227 + kx] * sw[oc * 49 + ky * 7 + kx];
    h1[k * 50176 + o] = fmaxf(acc, 0.f);
}

// classifier conv2 (16->32, 3x3, stride2, VALID -> 27x27) + relu + global mean
__global__ __launch_bounds__(256)
void k_cls_conv2_mean(const float* __restrict__ h1, const float* __restrict__ cw2,
                      float* __restrict__ gmean) {
    __shared__ float sw[144];
    __shared__ float red[256];
    int k = blockIdx.x, oc = blockIdx.y;
    for (int i = threadIdx.x; i < 144; i += 256) sw[i] = cw2[oc * 144 + i];
    __syncthreads();
    float part = 0.f;
    for (int o = threadIdx.x; o < 27 * 27; o += 256) {
        int oy = o / 27, ox = o - oy * 27;
        const float* hp = h1 + k * 50176 + (oy * 2) * 56 + ox * 2;
        float acc = 0.f;
        #pragma unroll
        for (int ic = 0; ic < 16; ++ic) {
            const float* hc = hp + ic * 3136;
            #pragma unroll
            for (int ky = 0; ky < 3; ++ky)
                #pragma unroll
                for (int kx = 0; kx < 3; ++kx)
                    acc += hc[ky * 56 + kx] * sw[ic * 9 + ky * 3 + kx];
        }
        part += fmaxf(acc, 0.f);
    }
    red[threadIdx.x] = part;
    __syncthreads();
    for (int s = 128; s > 0; s >>= 1) {
        if (threadIdx.x < s) red[threadIdx.x] += red[threadIdx.x + s];
        __syncthreads();
    }
    if (threadIdx.x == 0) gmean[k * 32 + oc] = red[0] / 729.f;
}

// logits = gmean @ fc ; argmax + validity (count >= MIN_SIZE)
__global__ void k_logits(const float* __restrict__ gmean, const float* __restrict__ fc,
                         const int* __restrict__ cnts, int* __restrict__ cls) {
    __shared__ float lg[16][4];
    int t = threadIdx.x;
    if (t < 64) {
        int k = t >> 2, c = t & 3;
        float acc = 0.f;
        for (int ic = 0; ic < 32; ++ic) acc += gmean[k * 32 + ic] * fc[ic * 4 + c];
        lg[k][c] = acc;
    }
    __syncthreads();
    if (t < 16) {
        int best = 0;
        float bv = lg[t][0];
        for (int c = 1; c < 4; ++c) if (lg[t][c] > bv) { bv = lg[t][c]; best = c; }
        cls[t] = (cnts[t] >= 64) ? best : -1;
    }
}

// final masks: per pixel find its component's class, write one-hot over 4 ch
__global__ __launch_bounds__(256)
void k_final(const int* __restrict__ lab, const int* __restrict__ lids,
             const int* __restrict__ cls, float* __restrict__ out) {
    __shared__ int slid[16], scls[16];
    int tid = threadIdx.x;
    if (tid < 16) { slid[tid] = lids[tid]; scls[tid] = cls[tid]; }
    __syncthreads();
    int p = blockIdx.x * 256 + tid;
    int v = lab[p];
    int cp = -1;
    if (v != BIGL) {
        #pragma unroll
        for (int k = 0; k < 16; ++k) if (slid[k] == v) cp = scls[k];
    }
    #pragma unroll
    for (int c = 0; c < 4; ++c) out[(size_t)c * BIGL + p] = (c == cp) ? 1.f : 0.f;
}

// ---------------------------------------------------------------------------
extern "C" void kernel_launch(void* const* d_in, const int* in_sizes, int n_in,
                              void* d_out, int out_size, void* d_ws, size_t ws_size,
                              hipStream_t stream) {
    (void)in_sizes; (void)n_in; (void)out_size; (void)ws_size;
    const float* x   = (const float*)d_in[0];
    const float* w1  = (const float*)d_in[1];
    const float* w2  = (const float*)d_in[2];
    const float* w3  = (const float*)d_in[3];
    const float* cw1 = (const float*)d_in[4];
    const float* cw2 = (const float*)d_in[5];
    const float* fc  = (const float*)d_in[6];
    float* out = (float*)d_out;

    char* ws = (char*)d_ws;
    size_t off = 0;
    auto take = [&](size_t bytes) -> void* {
        void* p = ws + off;
        off += (bytes + 255) & ~(size_t)255;
        return p;
    };
    int* labA    = (int*)take((size_t)BIGL * 4);
    int* labB    = (int*)take((size_t)BIGL * 4);
    int* counts  = (int*)take(((size_t)BIGL + 1) * 4);
    unsigned long long* partials = (unsigned long long*)take(1024 * 8);
    int* lids    = (int*)take(64);
    int* cnts    = (int*)take(64);
    int* bbox    = (int*)take(256);
    int* cls     = (int*)take(64);
    float* gmean = (float*)take(16 * 32 * 4);
    _Float16* wBuf = (_Float16*)take(9216 * 2);
    float* roi   = (float*)take((size_t)16 * 227 * 227 * 4);
    float* h1    = (float*)take((size_t)16 * 50176 * 4);

    k_prep_w2<<<36, 256, 0, stream>>>(w2, wBuf);
    k_cascade<<<dim3(128, 128), 256, 0, stream>>>(x, w1, w3, wBuf, labA);

    int* a = labA; int* b = labB;
    for (int it = 0; it < 64; ++it) {              // exactly CC_ITERS sweeps
        k_cc_step<<<16384, 256, 0, stream>>>(a, b);
        int* t = a; a = b; b = t;                  // even count -> ends in labA
    }

    k_fill_i32<<<16385, 256, 0, stream>>>(counts, BIGL + 1, 0);
    k_histo<<<1024, 256, 0, stream>>>(a, counts);

    for (int k = 0; k < 16; ++k) {
        k_amax_a<<<1024, 256, 0, stream>>>(counts, partials);
        k_amax_b<<<1, 256, 0, stream>>>(partials, counts, lids, cnts, k);
    }

    k_bbox_init<<<1, 64, 0, stream>>>(bbox);
    k_bbox_scan<<<16384, 256, 0, stream>>>(a, lids, bbox);

    k_roi<<<dim3(16, 202), 256, 0, stream>>>(x, bbox, roi);
    k_cls_conv1<<<dim3(16, 196), 256, 0, stream>>>(roi, cw1, h1);
    k_cls_conv2_mean<<<dim3(16, 32), 256, 0, stream>>>(h1, cw2, gmean);
    k_logits<<<1, 256, 0, stream>>>(gmean, fc, cnts, cls);

    k_final<<<16384, 256, 0, stream>>>(a, lids, cls, out);
}
